// AttentionHead_11974368821865
// MI455X (gfx1250) — matrix operand order repaired and checked by reading, not compiled, not run on hardware
//
#include <hip/hip_runtime.h>
#include <hip/hip_bf16.h>
#include <stdint.h>

// ---------------------------------------------------------------------------
// Single-head causal attention for MI455X (gfx1250, wave32, WMMA + TDM).
//   B=4, S=4096, D_MODEL=1024, D_KEY=64
// Pipeline:
//   proj_kernel x3 : Q/K/V = X @ W^T  (fp32 in -> bf16 out, f32 WMMA accum)
//   attn_kernel    : flash attention; K tiles staged by the Tensor Data Mover
//                    (tensor_load_to_lds + s_wait_tensorcnt), bf16 WMMA.
// ---------------------------------------------------------------------------

constexpr int SEQ    = 4096;
constexpr int NB     = 4;
constexpr int DMODEL = 1024;
constexpr int DKEY   = 64;

typedef __attribute__((ext_vector_type(8)))  float        v8f;
typedef __attribute__((ext_vector_type(16))) __bf16       v16bf;
typedef __attribute__((ext_vector_type(8)))  __bf16       bf16x8;
typedef __attribute__((ext_vector_type(4)))  unsigned int u32x4;
typedef __attribute__((ext_vector_type(8)))  int          i32x8;
typedef __attribute__((ext_vector_type(4)))  int          i32x4;

#if defined(__has_builtin)
#if __has_builtin(__builtin_amdgcn_tensor_load_to_lds) && \
    __has_builtin(__builtin_amdgcn_s_wait_tensorcnt)
#define USE_TDM 1
#endif
#endif
#ifndef USE_TDM
#define USE_TDM 0
#endif

// round-to-nearest-even f32 -> bf16
static __device__ __forceinline__ __bf16 f2bf(float f) {
    uint32_t u = __builtin_bit_cast(uint32_t, f);
    u += 0x7FFFu + ((u >> 16) & 1u);
    uint16_t h = (uint16_t)(u >> 16);
    return __builtin_bit_cast(__bf16, h);
}

// Build a 16-half WMMA fragment from two 16B-aligned 8-half chunks.
static __device__ __forceinline__ v16bf make_frag(const __bf16* p0, const __bf16* p1) {
    bf16x8 lo = *(const bf16x8*)p0;
    bf16x8 hi = *(const bf16x8*)p1;
    return __builtin_shufflevector(lo, hi, 0,1,2,3,4,5,6,7,8,9,10,11,12,13,14,15);
}

#if USE_TDM
// TDM: DMA a 64x64 bf16 tile (row pitch 64 elems) from global into LDS with
// hardware row padding so each LDS row is 72 halves (64 data + 8 pad = our
// bank-conflict-avoiding stride). D# packing per CDNA5 ISA
// sections 8.3/8.4: pad_interval=4 -> 32 DWORDs (one 64-bf16 row) between
// pads, pad_amount=3 -> 4 DWORDs (8 halves) of pad.
// This toolchain's builtin is the 6-arg form:
//   (uint32x4 g0, int32x8 g1, int32x4, int32x4, int32x8, i32 cpol)
// trailing groups zero-filled (2D tile: groups 2/3 unused, tile_dim2/3 = 0).
static __device__ __forceinline__ void tdm_load_tile_64x64_bf16(const __bf16* gsrc,
                                                                uint32_t lds_byte_off) {
    const uint64_t ga = (uint64_t)(uintptr_t)gsrc;
    u32x4 g0;
    g0.x = 1u;                                      // count=1 (valid user D#)
    g0.y = lds_byte_off;                            // lds_addr (bytes)
    g0.z = (uint32_t)(ga & 0xFFFFFFFFu);            // global_addr[31:0]
    g0.w = (uint32_t)((ga >> 32) & 0x01FFFFFFu)     // global_addr[56:32]
         | (2u << 30);                              // type = 2 ("image")
    i32x8 g1;
    g1[0] = (1 << 16)                               // data_size = 2 bytes
          | (1 << 20)                               // pad_enable
          | (4 << 22)                               // pad_interval: 32 DWORDs
          | (3 << 25);                              // pad_amount: 4 DWORDs
    g1[1] = (64 << 16);                             // tensor_dim0[15:0] = 64
    g1[2] = (64 << 16);                             // tensor_dim1[15:0] = 64
    g1[3] = (64 << 16);                             // tile_dim0 = 64
    g1[4] = 64;                                     // tile_dim1 = 64, tile_dim2 = 0
    g1[5] = 64;                                     // tensor_dim0_stride[31:0] = 64
    g1[6] = 0;                                      // dim0_stride hi, dim1_stride lo
    g1[7] = 0;                                      // dim1_stride hi
    const i32x4 z4 = {0, 0, 0, 0};
    const i32x8 z8 = {0, 0, 0, 0, 0, 0, 0, 0};
    __builtin_amdgcn_tensor_load_to_lds(g0, g1, z4, z4, z8, 0);
}
#endif

// ---------------------------------------------------------------------------
// Projection: OUT[b, s, n] (bf16) = sum_k X[b, s, k] * W[n, k]
// Block: 256 threads (8 waves): 128 S-rows x 64 outputs, K=1024 in chunks of
// 128 staged through LDS (f32 -> bf16 on store). 128 WMMAs per wave.
// ---------------------------------------------------------------------------
__global__ __launch_bounds__(256) void proj_kernel(const float* __restrict__ X,
                                                   const float* __restrict__ W,
                                                   __bf16* __restrict__ OUT) {
    constexpr int KCH = 128;
    __shared__ __align__(16) __bf16 Xs[128][KCH + 8];   // stride 136 halves (16B-mult)
    __shared__ __align__(16) __bf16 Ws[DKEY][KCH + 8];

    const int b    = blockIdx.y;
    const int s0   = blockIdx.x * 128;
    const int tid  = threadIdx.x;
    const int wv   = tid >> 5;
    const int lane = tid & 31;
    const int ln   = lane & 15;      // N-lane / M-row index inside 16-lane half
    const int grp  = lane >> 4;      // lane group (K-offset select)
    const int m0   = wv * 16;

    v8f acc[4];
    #pragma unroll
    for (int t = 0; t < 4; ++t) acc[t] = (v8f){0,0,0,0,0,0,0,0};

    const float* Xbase = X + ((size_t)b * SEQ + s0) * DMODEL;

    for (int kc = 0; kc < DMODEL; kc += KCH) {
        __syncthreads();
        #pragma unroll
        for (int i = 0; i < 16; ++i) {              // X tile 128x128 f32
            int slot = tid + i * 256;
            int r = slot >> 5, c4 = slot & 31;
            const float4 f = *(const float4*)(Xbase + (size_t)r * DMODEL + kc + c4 * 4);
            __bf16* d = &Xs[r][c4 * 4];
            d[0] = f2bf(f.x); d[1] = f2bf(f.y); d[2] = f2bf(f.z); d[3] = f2bf(f.w);
        }
        #pragma unroll
        for (int i = 0; i < 8; ++i) {               // W tile 64x128 f32
            int slot = tid + i * 256;
            int r = slot >> 5, c4 = slot & 31;
            const float4 f = *(const float4*)(W + (size_t)r * DMODEL + kc + c4 * 4);
            __bf16* d = &Ws[r][c4 * 4];
            d[0] = f2bf(f.x); d[1] = f2bf(f.y); d[2] = f2bf(f.z); d[3] = f2bf(f.w);
        }
        __syncthreads();

        #pragma unroll
        for (int kk = 0; kk < KCH; kk += 32) {
            const __bf16* arow = &Xs[m0 + ln][0];
            v16bf a = make_frag(arow + kk + grp * 8, arow + kk + 16 + grp * 8);
            #pragma unroll
            for (int t = 0; t < 4; ++t) {
                const __bf16* brow = &Ws[t * 16 + ln][0];
                v16bf bf = make_frag(brow + kk + grp * 8, brow + kk + 16 + grp * 8);
                acc[t] = __builtin_amdgcn_wmma_f32_16x16x32_bf16(
                    false, a, false, bf, (short)0, acc[t], false, false);
            }
        }
    }

    #pragma unroll
    for (int t = 0; t < 4; ++t)
        #pragma unroll
        for (int r = 0; r < 8; ++r) {
            int m = m0 + r + grp * 8;
            int n = t * 16 + ln;
            OUT[((size_t)b * SEQ + s0 + m) * DKEY + n] = f2bf(acc[t][r]);
        }
}

// ---------------------------------------------------------------------------
// Flash attention: 128 q-rows per block (8 waves x 16), 64-wide KV tiles.
// K tile staged by TDM (async DMA -> LDS with hw padding); V staged manually
// transposed so P*V B-fragments are vectorized 16B ds loads. Online softmax
// in registers matching the WMMA C-layout (shfl_xor row reductions).
// ---------------------------------------------------------------------------
__global__ __launch_bounds__(256) void attn_kernel(const __bf16* __restrict__ Q,
                                                   const __bf16* __restrict__ K,
                                                   const __bf16* __restrict__ V,
                                                   float* __restrict__ OUT) {
    __shared__ __align__(16) __bf16 Ks[64][72];       // K tile [s][d] (TDM-padded rows)
    __shared__ __align__(16) __bf16 Vt[64][72];       // V tile transposed [d][s]
    __shared__ __align__(16) __bf16 Ps[8][16][72];    // per-wave P scratch [m][s]

    const int b    = blockIdx.y;
    const int qb   = blockIdx.x * 128;
    const int tid  = threadIdx.x;
    const int wv   = tid >> 5;
    const int lane = tid & 31;
    const int ln   = lane & 15;
    const int grp  = lane >> 4;
    const int q0   = qb + wv * 16;

#if USE_TDM
    // gfx1250 generic LDS pointers carry the LDS byte offset in addr[31:0].
    const uint32_t ks_lds_off = (uint32_t)(uintptr_t)&Ks[0][0];
#endif

    // Q A-fragments for this wave's 16 rows (d = 0..31 and 32..63).
    const __bf16* qrow = Q + ((size_t)b * SEQ + q0 + ln) * DKEY;
    const v16bf aq0 = make_frag(qrow + grp * 8,      qrow + 16 + grp * 8);
    const v16bf aq1 = make_frag(qrow + 32 + grp * 8, qrow + 48 + grp * 8);

    v8f accO[4];
    #pragma unroll
    for (int t = 0; t < 4; ++t) accO[t] = (v8f){0,0,0,0,0,0,0,0};
    float m_st[8], l_st[8];
    #pragma unroll
    for (int r = 0; r < 8; ++r) { m_st[r] = -1.0e30f; l_st[r] = 0.0f; }

    const int ntiles = (qb + 128) / 64;   // causal: s <= qb+127
    for (int st = 0; st < ntiles; ++st) {
        const int sb = st * 64;
        __syncthreads();                  // previous tile fully consumed

#if USE_TDM
        if (wv == 0) {                    // one wave drives the DMA (EXEC ignored)
            tdm_load_tile_64x64_bf16(K + ((size_t)b * SEQ + sb) * DKEY, ks_lds_off);
        }
#else
        #pragma unroll
        for (int i = 0; i < 2; ++i) {
            int slot = tid + i * 256;
            int r = slot >> 3, c8 = slot & 7;
            bf16x8 kvv = *(const bf16x8*)(K + ((size_t)b * SEQ + sb + r) * DKEY + c8 * 8);
            *(bf16x8*)&Ks[r][c8 * 8] = kvv;
        }
#endif
        // V transpose staging overlaps with the TDM transfer.
        #pragma unroll
        for (int i = 0; i < 2; ++i) {
            int slot = tid + i * 256;                 // 512 slots of 8 halves
            int r = slot >> 3, c8 = slot & 7;
            bf16x8 vv = *(const bf16x8*)(V + ((size_t)b * SEQ + sb + r) * DKEY + c8 * 8);
            #pragma unroll
            for (int j = 0; j < 8; ++j) Vt[c8 * 8 + j][r] = vv[j];
        }
#if USE_TDM
        if (wv == 0) __builtin_amdgcn_s_wait_tensorcnt(0);
#endif
        __syncthreads();                  // K (TDM) + V tiles visible to all waves
        if (st + 1 < ntiles) {            // warm L2 for the next tile
            __builtin_prefetch(K + ((size_t)b * SEQ + sb + 64) * DKEY, 0, 1);
            __builtin_prefetch(V + ((size_t)b * SEQ + sb + 64) * DKEY, 0, 1);
        }

        // S = Q K^T : 4 s-subtiles x (K=64 -> 2 WMMAs)
        v8f sacc[4];
        #pragma unroll
        for (int t = 0; t < 4; ++t) {
            sacc[t] = (v8f){0,0,0,0,0,0,0,0};
            const __bf16* brow = &Ks[t * 16 + ln][0];
            v16bf b0 = make_frag(brow + grp * 8,      brow + 16 + grp * 8);
            v16bf b1 = make_frag(brow + 32 + grp * 8, brow + 48 + grp * 8);
            sacc[t] = __builtin_amdgcn_wmma_f32_16x16x32_bf16(
                false, aq0, false, b0, (short)0, sacc[t], false, false);
            sacc[t] = __builtin_amdgcn_wmma_f32_16x16x32_bf16(
                false, aq1, false, b1, (short)0, sacc[t], false, false);
        }

        // scale + causal mask + online softmax (per q-row r in C-layout)
        #pragma unroll
        for (int r = 0; r < 8; ++r) {
            const int qg = q0 + r + grp * 8;
            float rmax = -1.0e30f;
            #pragma unroll
            for (int t = 0; t < 4; ++t) {
                float x = sacc[t][r] * 0.125f;        // 1/sqrt(64)
                const int sg = sb + t * 16 + ln;
                if (sg > qg) x = -1.0e30f;            // causal mask
                sacc[t][r] = x;
                rmax = fmaxf(rmax, x);
            }
            rmax = fmaxf(rmax, __shfl_xor(rmax, 1, 16));
            rmax = fmaxf(rmax, __shfl_xor(rmax, 2, 16));
            rmax = fmaxf(rmax, __shfl_xor(rmax, 4, 16));
            rmax = fmaxf(rmax, __shfl_xor(rmax, 8, 16));
            const float mnew  = fmaxf(m_st[r], rmax);
            const float alpha = __expf(m_st[r] - mnew);
            float rsum = 0.0f;
            #pragma unroll
            for (int t = 0; t < 4; ++t) {
                float p = __expf(sacc[t][r] - mnew);
                sacc[t][r] = p;
                rsum += p;
            }
            rsum += __shfl_xor(rsum, 1, 16);
            rsum += __shfl_xor(rsum, 2, 16);
            rsum += __shfl_xor(rsum, 4, 16);
            rsum += __shfl_xor(rsum, 8, 16);
            l_st[r] = l_st[r] * alpha + rsum;
            m_st[r] = mnew;
            #pragma unroll
            for (int t = 0; t < 4; ++t) accO[t][r] *= alpha;
            #pragma unroll
            for (int t = 0; t < 4; ++t)               // spill P to per-wave LDS
                Ps[wv][r + grp * 8][t * 16 + ln] = f2bf(sacc[t][r]);
        }
        // Same-wave DS ops are in-order: no barrier needed before reload.

        // O += P V : A = P (16x64), B-fragments from transposed V tile
        const __bf16* prow = &Ps[wv][ln][0];
        v16bf pa0 = make_frag(prow + grp * 8,      prow + 16 + grp * 8);
        v16bf pa1 = make_frag(prow + 32 + grp * 8, prow + 48 + grp * 8);
        #pragma unroll
        for (int t = 0; t < 4; ++t) {
            const __bf16* vrow = &Vt[t * 16 + ln][0];
            v16bf bv0 = make_frag(vrow + grp * 8,      vrow + 16 + grp * 8);
            v16bf bv1 = make_frag(vrow + 32 + grp * 8, vrow + 48 + grp * 8);
            accO[t] = __builtin_amdgcn_wmma_f32_16x16x32_bf16(
                false, pa0, false, bv0, (short)0, accO[t], false, false);
            accO[t] = __builtin_amdgcn_wmma_f32_16x16x32_bf16(
                false, pa1, false, bv1, (short)0, accO[t], false, false);
        }
    }

    // epilogue: O / l  -> fp32 output [B, S, 64]
    #pragma unroll
    for (int t = 0; t < 4; ++t)
        #pragma unroll
        for (int r = 0; r < 8; ++r) {
            const int qg = q0 + r + grp * 8;
            OUT[((size_t)b * SEQ + qg) * DKEY + t * 16 + ln] = accO[t][r] / l_st[r];
        }
}

// ---------------------------------------------------------------------------
extern "C" void kernel_launch(void* const* d_in, const int* in_sizes, int n_in,
                              void* d_out, int out_size, void* d_ws, size_t ws_size,
                              hipStream_t stream) {
    (void)in_sizes; (void)n_in; (void)out_size; (void)ws_size;
    const float* queries = (const float*)d_in[0];
    const float* keys    = (const float*)d_in[1];
    const float* values  = (const float*)d_in[2];
    const float* Wq      = (const float*)d_in[3];
    const float* Wk      = (const float*)d_in[4];
    const float* Wv      = (const float*)d_in[5];
    // d_in[6] (mask) unused: causality computed analytically.

    const size_t np = (size_t)NB * SEQ * DKEY;       // bf16 Q/K/V: 2 MB each
    __bf16* qp = (__bf16*)d_ws;
    __bf16* kp = qp + np;
    __bf16* vp = kp + np;

    dim3 grid(SEQ / 128, NB);
    proj_kernel<<<grid, 256, 0, stream>>>(queries, Wq, qp);
    proj_kernel<<<grid, 256, 0, stream>>>(keys,    Wk, kp);
    proj_kernel<<<grid, 256, 0, stream>>>(values,  Wv, vp);
    attn_kernel<<<grid, 256, 0, stream>>>(qp, kp, vp, (float*)d_out);
}
